// WorkingMemory_8581344657475
// MI455X (gfx1250) — matrix operand order repaired and checked by reading, not compiled, not run on hardware
//
#include <hip/hip_runtime.h>
#include <math.h>

#define D   2048
#define TD  6144   // 3*D
#define B   512
#define NS  64
#define HD  1024   // D/2

typedef __attribute__((ext_vector_type(16))) __bf16 v16bf;
typedef __attribute__((ext_vector_type(8)))  float  v8f;

__device__ __forceinline__ float sigmoidf_(float x) { return 1.f / (1.f + expf(-x)); }

// =====================================================================
// WMMA bf16 GEMM:  C[M,N] = A[M,K] @ W[N,K]^T (+ bias[N])
// fp32 tiles staged in LDS via GLOBAL_LOAD_ASYNC_TO_LDS_B128 (ASYNCcnt),
// converted to bf16 while building v_wmma_f32_16x16x32_bf16 fragments.
// Block = 256 threads (8 wave32), tile 64x64, K-step 32, double-buffered.
// =====================================================================
#define BM 64
#define BN 64
#define BK 32

__global__ void __launch_bounds__(256)
k_gemm_bf16(const float* __restrict__ A, const float* __restrict__ W,
            const float* __restrict__ bias, float* __restrict__ C,
            int M, int N, int K)
{
  (void)M;
  __shared__ float lA[2][BM][BK];
  __shared__ float lW[2][BN][BK];

  const int tid  = threadIdx.x;
  const int lane = tid & 31;
  const int wave = tid >> 5;
  const int wm   = wave & 3;   // M subtile (16 rows)
  const int wn   = wave >> 2;  // N group (32 cols -> 2 WMMA tiles)
  const int bm0  = blockIdx.y * BM;
  const int bn0  = blockIdx.x * BN;

  // tile loader: 64x32 fp32 = 512 float4 segments; 2 per thread
  const int seg = tid * 2;
  const int r0 = seg >> 3,       c0 = (seg & 7) * 4;
  const int r1 = (seg + 1) >> 3, c1 = ((seg + 1) & 7) * 4;

  v8f acc0 = {};
  v8f acc1 = {};

  const int nk = K / BK;

  // gfx1250 async copy global->LDS (tracked by ASYNCcnt)
  #define ISSUE(buf, k0)                                                          \
    do {                                                                          \
      const float* gA0 = A + (size_t)(bm0 + r0) * K + (k0) + c0;                  \
      const float* gA1 = A + (size_t)(bm0 + r1) * K + (k0) + c1;                  \
      const float* gW0 = W + (size_t)(bn0 + r0) * K + (k0) + c0;                  \
      const float* gW1 = W + (size_t)(bn0 + r1) * K + (k0) + c1;                  \
      unsigned dA0 = (unsigned)(unsigned long long)&lA[buf][r0][c0];              \
      unsigned dA1 = (unsigned)(unsigned long long)&lA[buf][r1][c1];              \
      unsigned dW0 = (unsigned)(unsigned long long)&lW[buf][r0][c0];              \
      unsigned dW1 = (unsigned)(unsigned long long)&lW[buf][r1][c1];              \
      asm volatile("global_load_async_to_lds_b128 %0, %1, off" :: "v"(dA0), "v"(gA0) : "memory"); \
      asm volatile("global_load_async_to_lds_b128 %0, %1, off" :: "v"(dA1), "v"(gA1) : "memory"); \
      asm volatile("global_load_async_to_lds_b128 %0, %1, off" :: "v"(dW0), "v"(gW0) : "memory"); \
      asm volatile("global_load_async_to_lds_b128 %0, %1, off" :: "v"(dW1), "v"(gW1) : "memory"); \
    } while (0)

  ISSUE(0, 0);
  asm volatile("s_wait_asynccnt 0" ::: "memory");
  __syncthreads();

  for (int kt = 0; kt < nk; ++kt) {
    const int buf = kt & 1;
    if (kt + 1 < nk) { ISSUE(buf ^ 1, (kt + 1) * BK); }

    // A fragment 16x32 bf16: lane 0-15 -> K 0-15 of row lane;
    //                        lane 16-31 -> K 16-31 of row lane-16.
    const int ar = wm * 16 + (lane & 15);
    const int ka = (lane >> 4) * 16;
    v16bf af;
    #pragma unroll
    for (int j = 0; j < 8; ++j) {
      af[j]     = (__bf16)lA[buf][ar][ka + j];
      af[8 + j] = (__bf16)lA[buf][ar][ka + 8 + j];
    }
    // B fragments (W rows = columns of B=W^T): lane owns one N row,
    // 16 contiguous K values selected by lane half.
    const int kb  = (lane >> 4) * 16;
    const int wr0 = wn * 32 + (lane & 15);
    v16bf bf0;
    #pragma unroll
    for (int j = 0; j < 16; ++j) bf0[j] = (__bf16)lW[buf][wr0][kb + j];
    acc0 = __builtin_amdgcn_wmma_f32_16x16x32_bf16(false, af, false, bf0,
                                                   (short)0, acc0, false, false);
    const int wr1 = wr0 + 16;
    v16bf bf1;
    #pragma unroll
    for (int j = 0; j < 16; ++j) bf1[j] = (__bf16)lW[buf][wr1][kb + j];
    acc1 = __builtin_amdgcn_wmma_f32_16x16x32_bf16(false, af, false, bf1,
                                                   (short)0, acc1, false, false);

    asm volatile("s_wait_asynccnt 0" ::: "memory");
    __syncthreads();
  }
  #undef ISSUE

  // C layout: lane 0-15: N=lane, M=r; lane 16-31: N=lane-16, M=8+r
  const int rowB = bm0 + wm * 16 + 8 * (lane >> 4);
  const int colN = bn0 + wn * 32 + (lane & 15);
  #pragma unroll
  for (int r = 0; r < 8; ++r) {
    const int row = rowB + r;
    const float b0 = bias ? bias[colN]      : 0.f;
    const float b1 = bias ? bias[colN + 16] : 0.f;
    C[(size_t)row * N + colN]      = acc0[r] + b0;
    C[(size_t)row * N + colN + 16] = acc1[r] + b1;
  }
}

// ============================ small kernels ============================

__global__ void k_hprev(const float* __restrict__ slots0, float* __restrict__ hp) {
  int d = blockIdx.x * blockDim.x + threadIdx.x;
  if (d >= D) return;
  float s = 0.f;
  for (int n = 0; n < NS; ++n) s += slots0[n * D + d];
  hp[d] = s * (1.f / NS);
}

// gh[j] = h_prev . w_hh[j,:] + b_hh[j]   (one block per output)
__global__ void __launch_bounds__(256)
k_gh(const float* __restrict__ hp, const float* __restrict__ w_hh,
     const float* __restrict__ b_hh, float* __restrict__ gh) {
  __shared__ float red[8];
  const int j = blockIdx.x, tid = threadIdx.x;
  float a = 0.f;
  for (int k = tid; k < D; k += 256) a += hp[k] * w_hh[(size_t)j * D + k];
  for (int o = 16; o > 0; o >>= 1) a += __shfl_down(a, o, 32);
  if ((tid & 31) == 0) red[tid >> 5] = a;
  __syncthreads();
  if (tid == 0) {
    float t = 0.f;
    for (int w = 0; w < 8; ++w) t += red[w];
    gh[j] = t + b_hh[j];
  }
}

__global__ void k_gru(const float* __restrict__ gi, const float* __restrict__ gh,
                      const float* __restrict__ hp, float* __restrict__ h) {
  int idx = blockIdx.x * blockDim.x + threadIdx.x;
  if (idx >= B * D) return;
  const int b = idx / D, d = idx - b * D;
  const float* g = gi + (size_t)b * TD;
  const float r = sigmoidf_(g[d] + gh[d]);
  const float z = sigmoidf_(g[D + d] + gh[D + d]);
  const float n = tanhf(g[2 * D + d] + r * gh[2 * D + d]);
  h[idx] = (1.f - z) * n + z * hp[d];
}

__global__ void __launch_bounds__(256)
k_gates(const float* __restrict__ s, const float* __restrict__ h,
        const float* __restrict__ wg_w, const float* __restrict__ wg_b,
        const float* __restrict__ eg_w, const float* __restrict__ eg_b,
        float* __restrict__ wg, float* __restrict__ eg) {
  __shared__ float red[16];
  const int b = blockIdx.x, tid = threadIdx.x;
  float aw = 0.f, ae = 0.f;
  for (int d = tid; d < D; d += 256) {
    const float sv = s[(size_t)b * D + d], hv = h[(size_t)b * D + d];
    aw += sv * wg_w[d] + hv * wg_w[D + d];
    ae += sv * eg_w[d] + hv * eg_w[D + d];
  }
  for (int o = 16; o > 0; o >>= 1) { aw += __shfl_down(aw, o, 32); ae += __shfl_down(ae, o, 32); }
  if ((tid & 31) == 0) { red[tid >> 5] = aw; red[8 + (tid >> 5)] = ae; }
  __syncthreads();
  if (tid == 0) {
    float tw = 0.f, te = 0.f;
    for (int w = 0; w < 8; ++w) { tw += red[w]; te += red[8 + w]; }
    wg[b] = sigmoidf_(tw + wg_b[0]);
    eg[b] = sigmoidf_(te + eg_b[0]);
  }
}

__global__ void k_zero1(float* p) { if (threadIdx.x == 0 && blockIdx.x == 0) p[0] = 0.f; }

// imp[b,n] = sigmoid( sum_k gelu((1-e)S0P[n,k] + w*HP[b,k] + b1[k]) * w2[k] + b2 )
// one wave32 per (b,n) pair; also accumulates global sum for imp.mean().
__global__ void __launch_bounds__(256)
k_imp(const float* __restrict__ S0P, const float* __restrict__ HP,
      const float* __restrict__ imp_b1, const float* __restrict__ imp_w2,
      const float* __restrict__ imp_b2, const float* __restrict__ wg,
      const float* __restrict__ eg, float* __restrict__ imp,
      float* __restrict__ impsum) {
  const int lane = threadIdx.x & 31;
  const int pair = blockIdx.x * 8 + (threadIdx.x >> 5);
  const int b = pair >> 6, n = pair & 63;
  const float w = wg[b], ome = 1.f - eg[b];
  float acc = 0.f;
  for (int k = lane; k < HD; k += 32) {
    const float pre = ome * S0P[n * HD + k] + w * HP[(size_t)b * HD + k] + imp_b1[k];
    const float g = 0.5f * pre * (1.f + erff(pre * 0.70710678118f)); // exact gelu
    acc += g * imp_w2[k];
  }
  for (int o = 16; o > 0; o >>= 1) acc += __shfl_down(acc, o, 32);
  if (lane == 0) {
    const float v = sigmoidf_(acc + imp_b2[0]);
    imp[pair] = v;
    atomicAdd(impsum, v);
  }
}

// scatter flags: if imp.mean() > 0.9, mark the n_slots/4 lowest-imp slots
// (stable argsort tie-break by index) — those slots become exactly h[b,:].
__global__ void __launch_bounds__(64)
k_flags(const float* __restrict__ imp, const float* __restrict__ impsum,
        float* __restrict__ flags) {
  __shared__ float sh[NS];
  const int b = blockIdx.x, n = threadIdx.x;
  sh[n] = imp[b * NS + n];
  __syncthreads();
  float f = 0.f;
  if (impsum[0] * (1.f / (B * NS)) > 0.9f) {
    const float v = sh[n];
    int cnt = 0;
    for (int m = 0; m < NS; ++m) cnt += (sh[m] < v) || (sh[m] == v && m < n);
    f = (cnt < NS / 4) ? 1.f : 0.f;
  }
  flags[b * NS + n] = f;
}

// qh[b] = q[b].HSP[b] ; qb[b] = q[b].sp_b
__global__ void __launch_bounds__(256)
k_dots(const float* __restrict__ q, const float* __restrict__ HSP,
       const float* __restrict__ sp_b, float* __restrict__ qh, float* __restrict__ qb) {
  __shared__ float red[16];
  const int b = blockIdx.x, tid = threadIdx.x;
  float ah = 0.f, ab = 0.f;
  for (int d = tid; d < D; d += 256) {
    const float qv = q[(size_t)b * D + d];
    ah += qv * HSP[(size_t)b * D + d];
    ab += qv * sp_b[d];
  }
  for (int o = 16; o > 0; o >>= 1) { ah += __shfl_down(ah, o, 32); ab += __shfl_down(ab, o, 32); }
  if ((tid & 31) == 0) { red[tid >> 5] = ah; red[8 + (tid >> 5)] = ab; }
  __syncthreads();
  if (tid == 0) {
    float th = 0.f, tb = 0.f;
    for (int w = 0; w < 8; ++w) { th += red[w]; tb += red[8 + w]; }
    qh[b] = th; qb[b] = tb;
  }
}

// scores -> softmax -> attn_masked (for attn@slots0) and coef (h coefficient)
__global__ void __launch_bounds__(64)
k_softmax(const float* __restrict__ QS0, const float* __restrict__ qh,
          const float* __restrict__ qb, const float* __restrict__ wg,
          const float* __restrict__ eg, const float* __restrict__ flags,
          float* __restrict__ attnm, float* __restrict__ coef) {
  __shared__ float sb[NS];
  const int b = blockIdx.x, n = threadIdx.x;
  const float inv = 0.02209708691f; // 1/sqrt(2048)
  const float w = wg[b], e = eg[b], fl = flags[b * NS + n];
  const float sc = (fl > 0.5f)
      ? (qh[b] + qb[b]) * inv
      : ((1.f - e) * QS0[b * NS + n] + w * qh[b] + qb[b]) * inv;
  sb[n] = sc; __syncthreads();
  for (int s2 = 32; s2 > 0; s2 >>= 1) { if (n < s2) sb[n] = fmaxf(sb[n], sb[n + s2]); __syncthreads(); }
  const float mx = sb[0]; __syncthreads();
  const float ex = expf(sc - mx);
  sb[n] = ex; __syncthreads();
  for (int s2 = 32; s2 > 0; s2 >>= 1) { if (n < s2) sb[n] += sb[n + s2]; __syncthreads(); }
  const float attn = ex / sb[0]; __syncthreads();
  attnm[b * NS + n] = attn * (1.f - fl) * (1.f - e);
  sb[n] = (fl > 0.5f) ? attn : w * attn; __syncthreads();
  for (int s2 = 32; s2 > 0; s2 >>= 1) { if (n < s2) sb[n] += sb[n + s2]; __syncthreads(); }
  if (n == 0) coef[b] = sb[0];
}

// x = (1+coef)*h + attnm @ slots0 ; RMS norm ; scale by norm_w
__global__ void __launch_bounds__(256)
k_final(const float* __restrict__ h, const float* __restrict__ slots0,
        const float* __restrict__ attnm, const float* __restrict__ coef,
        const float* __restrict__ norm_w, float* __restrict__ out) {
  __shared__ float s_at[NS];
  __shared__ float red[9];
  const int b = blockIdx.x, tid = threadIdx.x;
  if (tid < NS) s_at[tid] = attnm[b * NS + tid];
  __syncthreads();
  const float cf = coef[b];
  float x[8]; float ss = 0.f;
  #pragma unroll
  for (int i = 0; i < 8; ++i) {
    const int d = tid + 256 * i;
    float acc = (1.f + cf) * h[(size_t)b * D + d];
    for (int n = 0; n < NS; ++n) acc += s_at[n] * slots0[n * D + d];
    x[i] = acc; ss += acc * acc;
  }
  for (int o = 16; o > 0; o >>= 1) ss += __shfl_down(ss, o, 32);
  if ((tid & 31) == 0) red[tid >> 5] = ss;
  __syncthreads();
  if (tid == 0) {
    float t = 0.f;
    for (int w = 0; w < 8; ++w) t += red[w];
    red[8] = rsqrtf(t * (1.f / D) + 1e-6f);
  }
  __syncthreads();
  const float rms = red[8];
  #pragma unroll
  for (int i = 0; i < 8; ++i) {
    const int d = tid + 256 * i;
    out[(size_t)b * D + d] = x[i] * rms * norm_w[d];
  }
}

// ============================ launch ============================

extern "C" void kernel_launch(void* const* d_in, const int* in_sizes, int n_in,
                              void* d_out, int out_size, void* d_ws, size_t ws_size,
                              hipStream_t stream) {
  (void)in_sizes; (void)n_in; (void)out_size; (void)ws_size;
  const float* s      = (const float*)d_in[0];
  const float* slots0 = (const float*)d_in[1];
  const float* w_ih   = (const float*)d_in[2];
  const float* w_hh   = (const float*)d_in[3];
  const float* b_ih   = (const float*)d_in[4];
  const float* b_hh   = (const float*)d_in[5];
  const float* wg_w   = (const float*)d_in[6];
  const float* wg_b   = (const float*)d_in[7];
  const float* eg_w   = (const float*)d_in[8];
  const float* eg_b   = (const float*)d_in[9];
  const float* imp_w1 = (const float*)d_in[10];
  const float* imp_b1 = (const float*)d_in[11];
  const float* imp_w2 = (const float*)d_in[12];
  const float* imp_b2 = (const float*)d_in[13];
  const float* q_w    = (const float*)d_in[14];
  const float* q_b    = (const float*)d_in[15];
  const float* sp_w   = (const float*)d_in[16];
  const float* sp_b   = (const float*)d_in[17];
  const float* norm_w = (const float*)d_in[18];
  float* out = (float*)d_out;

  float* ws = (float*)d_ws;
  float* hp     = ws; ws += D;
  float* gh     = ws; ws += TD;
  float* gi     = ws; ws += (size_t)B * TD;
  float* h      = ws; ws += (size_t)B * D;
  float* wg     = ws; ws += B;
  float* eg     = ws; ws += B;
  float* S0P    = ws; ws += NS * HD;
  float* HP     = ws; ws += (size_t)B * HD;
  float* imp    = ws; ws += B * NS;
  float* impsum = ws; ws += 1;
  float* flags  = ws; ws += B * NS;
  float* S0SP   = ws; ws += NS * D;
  float* HSP    = ws; ws += (size_t)B * D;
  float* q      = ws; ws += (size_t)B * D;
  float* QS0    = ws; ws += B * NS;
  float* qh     = ws; ws += B;
  float* qb     = ws; ws += B;
  float* attnm  = ws; ws += B * NS;
  float* coef   = ws; ws += B;

  k_hprev<<<dim3((D + 255) / 256), dim3(256), 0, stream>>>(slots0, hp);
  k_gh<<<dim3(TD), dim3(256), 0, stream>>>(hp, w_hh, b_hh, gh);
  // gi = s @ w_ih^T + b_ih  (512 x 6144, K=2048) — dominant GEMM
  k_gemm_bf16<<<dim3(TD / BN, B / BM), dim3(256), 0, stream>>>(s, w_ih, b_ih, gi, B, TD, D);
  k_gru<<<dim3((B * D + 255) / 256), dim3(256), 0, stream>>>(gi, gh, hp, h);
  k_gates<<<dim3(B), dim3(256), 0, stream>>>(s, h, wg_w, wg_b, eg_w, eg_b, wg, eg);
  // rank-factored importance path
  k_gemm_bf16<<<dim3(HD / BN, NS / BM), dim3(256), 0, stream>>>(slots0, imp_w1, nullptr, S0P, NS, HD, D);
  k_gemm_bf16<<<dim3(HD / BN, B / BM), dim3(256), 0, stream>>>(h, imp_w1, nullptr, HP, B, HD, D);
  k_zero1<<<1, 1, 0, stream>>>(impsum);
  k_imp<<<dim3(B * NS / 8), dim3(256), 0, stream>>>(S0P, HP, imp_b1, imp_w2, imp_b2, wg, eg, imp, impsum);
  k_flags<<<dim3(B), dim3(64), 0, stream>>>(imp, impsum, flags);
  // rank-factored attention path
  k_gemm_bf16<<<dim3(D / BN, B / BM), dim3(256), 0, stream>>>(h, q_w, q_b, q, B, D, D);
  k_gemm_bf16<<<dim3(D / BN, B / BM), dim3(256), 0, stream>>>(h, sp_w, nullptr, HSP, B, D, D);
  k_gemm_bf16<<<dim3(D / BN, NS / BM), dim3(256), 0, stream>>>(slots0, sp_w, nullptr, S0SP, NS, D, D);
  k_gemm_bf16<<<dim3(NS / BN, B / BM), dim3(256), 0, stream>>>(q, S0SP, nullptr, QS0, B, NS, D);
  k_dots<<<dim3(B), dim3(256), 0, stream>>>(q, HSP, sp_b, qh, qb);
  k_softmax<<<dim3(B), dim3(64), 0, stream>>>(QS0, qh, qb, wg, eg, flags, attnm, coef);
  k_final<<<dim3(B), dim3(256), 0, stream>>>(h, slots0, attnm, coef, norm_w, out);
}
